// TGAT_53824530153551
// MI455X (gfx1250) — matrix operand order repaired and checked by reading, not compiled, not run on hardware
//
#include <hip/hip_runtime.h>

// ---------------------------------------------------------------------------
// TGAT forward for MI455X (gfx1250), wave32, bf16 WMMA path.
//
// Shapes: B=1024, LQ=32, LK=64, H=4, D_MODEL=512, D_K=D_V=128.
//
// Pipeline:
//  1) convw   : Wv, fcW  -> bf16 copies
//  2) wmap    : wqh[h,:] = Wq[h*128:(h+1)*128,:]^T @ wm_q  (and wkh)  [exact]
//  3) sqsk    : sq[h,b,q] = q-row . wqh[h]; sk likewise               [exact fp32]
//  4) gemm_vp : vp = v @ Wv^T via v_wmma_f32_16x16x32_bf16, stored as
//               vpT[b][h][d][k] bf16 (transposed for step 5 B-operand)
//  5) attn    : per (b,h): logits = sq+sk, mask, softmax (fp32, written to
//               d_out), then WMMA attn(32x64) @ vp(64x128) -> ao bf16
//  6) gemm_fc : ao @ fcW^T via WMMA + bias + leaky_relu(0.2) + residual q
//               + LayerNorm(eps=1e-5) -> d_out
// ---------------------------------------------------------------------------

typedef __attribute__((ext_vector_type(16))) __bf16        v16bf;
typedef __attribute__((ext_vector_type(8)))  float         v8f;
typedef __attribute__((ext_vector_type(4)))  unsigned int  u32x4;
typedef __attribute__((ext_vector_type(4)))  float         f32x4;

union BF16x16 { u32x4 q[2]; v16bf v; };
union BF16x8  { __bf16 b[8]; u32x4 u; };

#define NB   1024
#define NLQ  32
#define NLK  64
#define NH   4
#define DMD  512

static __device__ inline v8f vzero8() {
    v8f z;
#pragma unroll
    for (int i = 0; i < 8; ++i) z[i] = 0.0f;
    return z;
}

// -------------------- 1) weight fp32 -> bf16 conversion ---------------------
__global__ __launch_bounds__(256) void convw_kernel(const float* __restrict__ Wv,
                                                    const float* __restrict__ fcW,
                                                    __bf16* __restrict__ Wvb,
                                                    __bf16* __restrict__ fcWb) {
    int i = blockIdx.x * 256 + threadIdx.x;   // 0..262143
    Wvb[i]  = (__bf16)Wv[i];
    fcWb[i] = (__bf16)fcW[i];
}

// -------------------- 2) collapse Wq/Wk through wm --------------------------
__global__ __launch_bounds__(256) void wmap_kernel(const float* __restrict__ Wq,
                                                   const float* __restrict__ Wk,
                                                   const float* __restrict__ Wmap,
                                                   float* __restrict__ wqh,
                                                   float* __restrict__ wkh) {
    int idx = blockIdx.x * 256 + threadIdx.x;     // 0..4095
    int mat = idx >> 11;                          // 0 = q-path, 1 = k-path
    int h   = (idx >> 9) & 3;
    int m   = idx & 511;
    const float* W  = mat ? Wk : Wq;
    const float* wm = Wmap + mat * 128;
    float acc = 0.0f;
    for (int d = 0; d < 128; ++d)
        acc += W[(long)(h * 128 + d) * DMD + m] * wm[d];
    (mat ? wkh : wqh)[h * DMD + m] = acc;
}

// -------------------- 3) sq / sk batched GEMV (wave-per-row) ----------------
__global__ __launch_bounds__(256) void sqsk_kernel(const float* __restrict__ qin,
                                                   const float* __restrict__ kin,
                                                   const float* __restrict__ wqh,
                                                   const float* __restrict__ wkh,
                                                   float* __restrict__ sqo,
                                                   float* __restrict__ sko) {
    const int w = threadIdx.x >> 5, L = threadIdx.x & 31;
    const long row = (long)blockIdx.x * 8 + w;    // 0..98303
    const bool isQ = (row < (long)NB * NLQ);
    const long  r  = isQ ? row : row - (long)NB * NLQ;
    const float* src = (isQ ? qin : kin) + r * DMD;
    const float* wt  = isQ ? wqh : wkh;
    float sv[16];
#pragma unroll
    for (int j = 0; j < 16; ++j) sv[j] = src[L + j * 32];
#pragma unroll
    for (int h = 0; h < 4; ++h) {
        const float* wh = wt + h * DMD;
        float p = 0.0f;
#pragma unroll
        for (int j = 0; j < 16; ++j) p += sv[j] * wh[L + j * 32];
#pragma unroll
        for (int off = 16; off > 0; off >>= 1) p += __shfl_xor(p, off, 32);
        if (L == 0) {
            if (isQ) sqo[(long)h * (NB * NLQ) + r] = p;
            else     sko[(long)h * (NB * NLK) + r] = p;
        }
    }
}

// -------------------- 4) vp = v @ Wv^T (bf16 WMMA), store transposed --------
// Block: 256 thr (8 waves). Block = 16 rows of (b,k); wave handles 4 N-tiles.
__global__ __launch_bounds__(256) void gemm_vp_kernel(const float* __restrict__ v,
                                                      const __bf16* __restrict__ Wvb,
                                                      __bf16* __restrict__ vpT) {
    const int tid = threadIdx.x;
    const int w = tid >> 5, L = tid & 31;
    const int nl = L & 15, half = L >> 4;
    const int m0 = blockIdx.x * 16;               // row strip in (b*64+k)
    const float* arow = v + (long)(m0 + nl) * DMD;

    v8f acc[4];
#pragma unroll
    for (int t = 0; t < 4; ++t) acc[t] = vzero8();

    for (int ks = 0; ks < 16; ++ks) {
        const int k0 = ks * 32;
        // A tile (fp32 -> bf16 on the fly): one contiguous 16-float chunk
        const float* p0 = arow + k0 + half * 16;
        f32x4 a0 = *(const f32x4*)(p0);
        f32x4 a1 = *(const f32x4*)(p0 + 4);
        f32x4 a2 = *(const f32x4*)(p0 + 8);
        f32x4 a3 = *(const f32x4*)(p0 + 12);
        BF16x16 A;
#pragma unroll
        for (int i = 0; i < 4; ++i) {
            A.v[i]      = (__bf16)a0[i];
            A.v[4 + i]  = (__bf16)a1[i];
            A.v[8 + i]  = (__bf16)a2[i];
            A.v[12 + i] = (__bf16)a3[i];
        }
#pragma unroll
        for (int t = 0; t < 4; ++t) {
            const int n = (w * 4 + t) * 16 + nl;
            v16bf Bv = *(const v16bf*)(Wvb + (long)n * DMD + k0 + half * 16);
            acc[t] = __builtin_amdgcn_wmma_f32_16x16x32_bf16(
                false, A.v, false, Bv, (short)0, acc[t], false, false);
        }
    }
    // Store: lane holds C[r+8*half][n] for r=0..7 -> 8 consecutive k in vpT
    const int b     = m0 >> 6;
    const int kbase = (m0 & 63) + half * 8;
#pragma unroll
    for (int t = 0; t < 4; ++t) {
        const int dg = (w * 4 + t) * 16 + nl;     // h*128 + d
        BF16x8 P;
#pragma unroll
        for (int r = 0; r < 8; ++r) P.b[r] = (__bf16)acc[t][r];
        *(u32x4*)(vpT + ((long)b * 512 + dg) * 64 + kbase) = P.u;
    }
}

// -------------------- 5) softmax + attn @ vp (per b,h; one wave) ------------
__global__ __launch_bounds__(32) void attn_kernel(const float* __restrict__ sq,
                                                  const float* __restrict__ sk,
                                                  const unsigned char* __restrict__ mask,
                                                  const __bf16* __restrict__ vpT,
                                                  float* __restrict__ attn_out,
                                                  __bf16* __restrict__ aob) {
    __shared__ float  sk_s[64];
    __shared__ __bf16 attn_s[32 * 72];    // row stride 72 bf16 (pad)
    __shared__ __bf16 ao_s[32 * 136];     // row stride 136 bf16 (pad)

    const int h = blockIdx.x >> 10;
    const int b = blockIdx.x & 1023;
    const int L = threadIdx.x;

    sk_s[L]      = sk[((long)h * NB + b) * 64 + L];
    sk_s[32 + L] = sk[((long)h * NB + b) * 64 + 32 + L];
    const float sqv = sq[((long)h * NB + b) * 32 + L];
    __syncthreads();

    // Softmax over this lane's row q = L
    const unsigned int* mrow = (const unsigned int*)(mask + ((long)b * 32 + L) * 64);
    float lg[64];
    float mx = -3.4e38f;
#pragma unroll
    for (int kk = 0; kk < 16; ++kk) {
        unsigned int mu = mrow[kk];
#pragma unroll
        for (int j = 0; j < 4; ++j) {
            int kki = kk * 4 + j;
            float val = ((mu >> (8 * j)) & 0xFF) ? -1.0e10f : (sqv + sk_s[kki]);
            lg[kki] = val;
            mx = fmaxf(mx, val);
        }
    }
    float ssum = 0.0f;
#pragma unroll
    for (int kki = 0; kki < 64; ++kki) {
        float e = __expf(lg[kki] - mx);
        lg[kki] = e;
        ssum += e;
    }
    const float inv = 1.0f / ssum;
    float* arow = attn_out + (((long)h * NB + b) * 32 + L) * 64;
#pragma unroll
    for (int kki = 0; kki < 64; ++kki) lg[kki] *= inv;
#pragma unroll
    for (int kki = 0; kki < 64; kki += 4) {
        f32x4 o = { lg[kki], lg[kki + 1], lg[kki + 2], lg[kki + 3] };
        *(f32x4*)(arow + kki) = o;
    }
#pragma unroll
    for (int kki = 0; kki < 64; ++kki) attn_s[L * 72 + kki] = (__bf16)lg[kki];
    __syncthreads();

    // WMMA: out(32x128) = attn(32x64) @ vp_bh(64x128)
    const int nl = L & 15, half = L >> 4;
    const __bf16* vbase = vpT + ((long)b * 512 + h * 128) * 64;
#pragma unroll
    for (int tm = 0; tm < 2; ++tm) {
#pragma unroll
        for (int tn = 0; tn < 8; ++tn) {
            v8f acc = vzero8();
#pragma unroll
            for (int ks = 0; ks < 2; ++ks) {
                const __bf16* ap = attn_s + (tm * 16 + nl) * 72 + ks * 32 + half * 16;
                BF16x16 A;
                A.q[0] = *(const u32x4*)(ap);
                A.q[1] = *(const u32x4*)(ap + 8);
                v16bf Bv = *(const v16bf*)(vbase + (long)(tn * 16 + nl) * 64 +
                                           ks * 32 + half * 16);
                acc = __builtin_amdgcn_wmma_f32_16x16x32_bf16(
                    false, A.v, false, Bv, (short)0, acc, false, false);
            }
#pragma unroll
            for (int r = 0; r < 8; ++r)
                ao_s[(tm * 16 + r + 8 * half) * 136 + tn * 16 + nl] = (__bf16)acc[r];
        }
    }
    __syncthreads();

    // Coalesced store of ao tile (32 x 128 bf16) -> ao[b][q][h*128 + d]
#pragma unroll
    for (int it = 0; it < 16; ++it) {
        int idx = it * 32 + L;                    // 512 chunks of 8 bf16
        int r   = idx >> 4;
        int c8  = (idx & 15) * 8;
        u32x4 d = *(const u32x4*)(ao_s + r * 136 + c8);
        *(u32x4*)(aob + ((long)(b * 32 + r)) * DMD + h * 128 + c8) = d;
    }
}

// ----- 6) fc = leaky_relu(ao @ fcW^T + fcb) + q; LayerNorm -> d_out ---------
// Block: 256 thr (8 waves), 16 output rows per block.
__global__ __launch_bounds__(256) void gemm_fc_ln_kernel(const __bf16* __restrict__ aob,
                                                         const __bf16* __restrict__ fcWb,
                                                         const float* __restrict__ fcb,
                                                         const float* __restrict__ qin,
                                                         const float* __restrict__ gamma,
                                                         const float* __restrict__ beta,
                                                         float* __restrict__ outp) {
    __shared__ float ldsq[16 * 516];
    __shared__ float ldso[16 * 516];
    const int tid = threadIdx.x;
    const int m0  = blockIdx.x * 16;              // rows of (b*32+q)

    for (int i = tid; i < 16 * 512; i += 256) {
        int r = i >> 9, c = i & 511;
        ldsq[r * 516 + c] = qin[(long)(m0 + r) * DMD + c];
    }
    __syncthreads();

    const int w = tid >> 5, L = tid & 31, nl = L & 15, half = L >> 4;
    v8f acc[4];
#pragma unroll
    for (int t = 0; t < 4; ++t) acc[t] = vzero8();

    const __bf16* arow = aob + (long)(m0 + nl) * DMD;
    for (int ks = 0; ks < 16; ++ks) {
        const int k0 = ks * 32;
        BF16x16 A;
        A.q[0] = *(const u32x4*)(arow + k0 + half * 16);
        A.q[1] = *(const u32x4*)(arow + k0 + 8 + half * 16);
#pragma unroll
        for (int t = 0; t < 4; ++t) {
            const int n = (w * 4 + t) * 16 + nl;
            v16bf Bv = *(const v16bf*)(fcWb + (long)n * DMD + k0 + half * 16);
            acc[t] = __builtin_amdgcn_wmma_f32_16x16x32_bf16(
                false, A.v, false, Bv, (short)0, acc[t], false, false);
        }
    }
    // bias + leaky relu + residual into LDS
#pragma unroll
    for (int t = 0; t < 4; ++t) {
        const int col = (w * 4 + t) * 16 + nl;
        const float bias = fcb[col];
#pragma unroll
        for (int r = 0; r < 8; ++r) {
            float x = acc[t][r] + bias;
            x = x > 0.0f ? x : 0.2f * x;
            int rl = r + 8 * half;
            x += ldsq[rl * 516 + col];
            ldso[rl * 516 + col] = x;
        }
    }
    __syncthreads();

    // LayerNorm: wave w handles rows w and w+8
    for (int rr = w; rr < 16; rr += 8) {
        float s = 0.0f, s2 = 0.0f;
        float xv[16];
#pragma unroll
        for (int j = 0; j < 16; ++j) {
            float x = ldso[rr * 516 + L * 16 + j];
            xv[j] = x; s += x; s2 += x * x;
        }
#pragma unroll
        for (int off = 16; off > 0; off >>= 1) {
            s  += __shfl_xor(s,  off, 32);
            s2 += __shfl_xor(s2, off, 32);
        }
        const float mu   = s * (1.0f / 512.0f);
        const float var  = s2 * (1.0f / 512.0f) - mu * mu;
        const float rstd = rsqrtf(var + 1e-5f);
        float* orow = outp + (long)(m0 + rr) * DMD;
#pragma unroll
        for (int j = 0; j < 16; ++j) {
            int c = L * 16 + j;
            orow[c] = gamma[c] * ((xv[j] - mu) * rstd) + beta[c];
        }
    }
}

// ---------------------------------------------------------------------------
extern "C" void kernel_launch(void* const* d_in, const int* in_sizes, int n_in,
                              void* d_out, int out_size, void* d_ws, size_t ws_size,
                              hipStream_t stream) {
    const float* q     = (const float*)d_in[0];
    const float* k     = (const float*)d_in[1];
    const float* v     = (const float*)d_in[2];
    const unsigned char* mask = (const unsigned char*)d_in[3];
    const float* Wq    = (const float*)d_in[4];
    const float* Wk    = (const float*)d_in[5];
    const float* Wv    = (const float*)d_in[6];
    const float* Wmap  = (const float*)d_in[7];
    const float* fcW   = (const float*)d_in[8];
    const float* fcb   = (const float*)d_in[9];
    const float* gamma = (const float*)d_in[10];
    const float* beta  = (const float*)d_in[11];

    char* ws = (char*)d_ws;
    __bf16* Wvb  = (__bf16*)(ws);                  //   524288 B
    __bf16* fcWb = (__bf16*)(ws + 524288);         //   524288 B
    float*  wqh  = (float*)(ws + 1048576);         //     8192 B
    float*  wkh  = (float*)(ws + 1056768);         //     8192 B
    float*  sq   = (float*)(ws + 1064960);         //   524288 B
    float*  sk   = (float*)(ws + 1589248);         //  1048576 B
    __bf16* vpT  = (__bf16*)(ws + 2637824);        // 67108864 B
    __bf16* aob  = (__bf16*)(ws + 69746688);       // 33554432 B  (end ~98.5 MB)

    float* outp     = (float*)d_out;
    float* attn_out = (float*)d_out + (long)NB * NLQ * DMD;

    convw_kernel<<<1024, 256, 0, stream>>>(Wv, fcW, Wvb, fcWb);
    wmap_kernel<<<16, 256, 0, stream>>>(Wq, Wk, Wmap, wqh, wkh);
    sqsk_kernel<<<12288, 256, 0, stream>>>(q, k, wqh, wkh, sq, sk);
    gemm_vp_kernel<<<4096, 256, 0, stream>>>(v, Wvb, vpT);
    attn_kernel<<<4096, 32, 0, stream>>>(sq, sk, mask, vpT, attn_out, aob);
    gemm_fc_ln_kernel<<<2048, 256, 0, stream>>>(aob, fcWb, fcb, q, gamma, beta, outp);
}
